// FeedForwardBackbone_61254823575552
// MI455X (gfx1250) — matrix operand order repaired and checked by reading, not compiled, not run on hardware
//
#include <hip/hip_runtime.h>
#include <hip/hip_bf16.h>

#define T_STEPS 128
#define BATCH   16
#define IN_DIM  1024
#define H_DIM   4096
#define OUT_DIM 1024
#define BETA    0.9f
#define THR     1.0f

#define NBLK 128
#define TPB  64     // 2 waves/block -> 256 waves == 256 column tiles of H

typedef __attribute__((ext_vector_type(16))) __bf16 v16bf;
typedef __attribute__((ext_vector_type(8)))  float  v8f;

union FragU { int4 q[2]; v16bf v; };

__device__ __forceinline__ v8f wmma_bf16(v16bf a, v16bf b, v8f c) {
  // (neg_a, A, neg_b, B, c_mod, C, reuse_a, reuse_b)
  return __builtin_amdgcn_wmma_f32_16x16x32_bf16(false, a, false, b, (short)0, c,
                                                 false, false);
}

// A fragment (16x32 bf16) gathered from a row-major [16][Kdim] bf16 matrix.
// Lane L: row = L&15, khalf = L>>4. Per-lane data = one contiguous 32B run (k = 16*khalf + s).
__device__ __forceinline__ v16bf load_afrag(const __hip_bfloat16* A, int Kdim,
                                            int kt, int l15, int khalf) {
  const char* p = (const char*)A + ((size_t)l15 * Kdim) * 2 + (size_t)kt * 64
                + khalf * 32;
  FragU f;
  f.q[0] = *(const int4*)p;
  f.q[1] = *(const int4*)(p + 16);
  return f.v;
}

// B fragment (32x16 bf16) from pre-packed layout: fragment = 512B, lane-major.
__device__ __forceinline__ v16bf load_bfrag(const __hip_bfloat16* base,
                                            long frag, int lane) {
  const char* p = (const char*)base + frag * 512 + (size_t)lane * 32;
  FragU f;
  f.q[0] = *(const int4*)p;
  f.q[1] = *(const int4*)(p + 16);
  return f.v;
}

// Monotonic device-wide barrier: counter never resets during the kernel;
// barrier r is complete when cnt >= r*NBLK.
__device__ __forceinline__ void grid_barrier(unsigned* cnt, unsigned target) {
  __threadfence();
  __syncthreads();
  if (threadIdx.x == 0) {
    __hip_atomic_fetch_add(cnt, 1u, __ATOMIC_ACQ_REL, __HIP_MEMORY_SCOPE_AGENT);
    while (__hip_atomic_load(cnt, __ATOMIC_ACQUIRE, __HIP_MEMORY_SCOPE_AGENT)
           < target) {
      __builtin_amdgcn_s_sleep(1);
    }
  }
  __syncthreads();
  __threadfence();
}

// ---------------- packing kernels (run once per launch, deterministic) -----

__global__ void init_bar(unsigned* bar) { bar[0] = 0u; }

// x [B][T][IN] f32 -> time-major bf16 hi/lo [T][B][IN]
__global__ void pack_x(const float* __restrict__ x,
                       __hip_bfloat16* __restrict__ hi,
                       __hip_bfloat16* __restrict__ lo) {
  size_t idx = (size_t)blockIdx.x * blockDim.x + threadIdx.x;
  if (idx >= (size_t)BATCH * T_STEPS * IN_DIM) return;
  int b   = (int)(idx >> 17);          // / (T*IN) = 131072
  int rem = (int)(idx & 131071);
  int t = rem >> 10;
  int i = rem & 1023;
  float a = x[idx];
  size_t dst = ((size_t)t * BATCH + b) * IN_DIM + i;
  __hip_bfloat16 h = __float2bfloat16(a);
  hi[dst] = h;
  lo[dst] = __float2bfloat16(a - __bfloat162float(h));
}

// w [N][K] f32 -> bf16 hi/lo in WMMA B-fragment layout.
// fragment f = nt*(K/32)+kt; dword slot = lane*8+j holds
// {w[nt*16+(lane&15)][kt*32+(lane>>4)*16+2j], w[..][..+1]}
__global__ void pack_w(const float* __restrict__ w,
                       __hip_bfloat16* __restrict__ hi,
                       __hip_bfloat16* __restrict__ lo, int N, int K) {
  long tid = (long)blockIdx.x * blockDim.x + threadIdx.x;
  long total = (long)N * K / 2;
  if (tid >= total) return;
  int  ktiles = K >> 5;
  long f    = tid >> 8;
  int  slot = (int)(tid & 255);
  int  lane = slot >> 3, j = slot & 7;
  int  nt = (int)(f / ktiles), kt = (int)(f % ktiles);
  int  n = nt * 16 + (lane & 15);
  int  k = kt * 32 + (lane >> 4) * 16 + j * 2;
  float a0 = w[(size_t)n * K + k];
  float a1 = w[(size_t)n * K + k + 1];
  __hip_bfloat16 h0 = __float2bfloat16(a0);
  __hip_bfloat16 h1 = __float2bfloat16(a1);
  hi[tid * 2]     = h0;
  hi[tid * 2 + 1] = h1;
  lo[tid * 2]     = __float2bfloat16(a0 - __bfloat162float(h0));
  lo[tid * 2 + 1] = __float2bfloat16(a1 - __bfloat162float(h1));
}

// ---------------- persistent SNN kernel -----------------------------------

__global__ __launch_bounds__(TPB, 1) void snn_persistent(
    const __hip_bfloat16* __restrict__ xb_hi, const __hip_bfloat16* __restrict__ xb_lo,
    const __hip_bfloat16* __restrict__ w1h,   const __hip_bfloat16* __restrict__ w1l,
    const __hip_bfloat16* __restrict__ w2h,   const __hip_bfloat16* __restrict__ w2l,
    const __hip_bfloat16* __restrict__ w3h,   const __hip_bfloat16* __restrict__ w3l,
    const float* __restrict__ b1, const float* __restrict__ b2,
    const float* __restrict__ b3,
    __hip_bfloat16* __restrict__ s1, __hip_bfloat16* __restrict__ acc2,
    unsigned* __restrict__ bar, float* __restrict__ out) {
  const int lane  = threadIdx.x & 31;
  const int wave  = blockIdx.x * (TPB >> 5) + (threadIdx.x >> 5); // 0..255
  const int l15   = lane & 15;
  const int khalf = lane >> 4;
  const int nt    = wave;                       // owns H columns [16nt,16nt+16)

  const float bias1 = b1[nt * 16 + l15];
  const float bias2 = b2[nt * 16 + l15];

  v8f v1, v2, spkcnt;
#pragma unroll
  for (int r = 0; r < 8; ++r) { v1[r] = 0.f; v2[r] = 0.f; spkcnt[r] = 0.f; }

  unsigned round = 0;

  for (int t = 0; t < T_STEPS; ++t) {
    // ---- layer 1: h1 = x_t @ W1^T + b1 --------------------------------
    // hi/lo split -> 3 independent accumulator chains (no WMMA->WMMA RAW
    // inside an iteration; dependent WMMAs are 3 apart).
    const __hip_bfloat16* Ah = xb_hi + (size_t)t * BATCH * IN_DIM;
    const __hip_bfloat16* Al = xb_lo + (size_t)t * BATCH * IN_DIM;
    v8f c_hh, c_hl, c_lh;
#pragma unroll
    for (int r = 0; r < 8; ++r) { c_hh[r] = bias1; c_hl[r] = 0.f; c_lh[r] = 0.f; }
    const long f1 = (long)nt * (IN_DIM / 32);
    for (int kt = 0; kt < IN_DIM / 32; ++kt) {
      v16bf ah = load_afrag(Ah, IN_DIM, kt, l15, khalf);
      v16bf al = load_afrag(Al, IN_DIM, kt, l15, khalf);
      v16bf bh = load_bfrag(w1h, f1 + kt, lane);
      v16bf bl = load_bfrag(w1l, f1 + kt, lane);
      c_hh = wmma_bf16(ah, bh, c_hh);
      c_hl = wmma_bf16(ah, bl, c_hl);
      c_lh = wmma_bf16(al, bh, c_lh);
    }
    v8f c = (c_hh + c_hl) + c_lh;
    // LIF 1 + publish spikes (exact in bf16)
#pragma unroll
    for (int r = 0; r < 8; ++r) {
      float v = BETA * v1[r] + c[r];
      float s = (v > THR) ? 1.0f : 0.0f;
      v1[r] = v - s * THR;
      s1[(size_t)(r + khalf * 8) * H_DIM + nt * 16 + l15] = __float2bfloat16(s);
    }
    ++round;
    grid_barrier(bar, round * NBLK);            // s1 ready for all readers

    // ---- layer 2: h2 = s1 @ W2^T + b2 (A exact; 2 independent chains) --
    v8f c2h, c2l;
#pragma unroll
    for (int r = 0; r < 8; ++r) { c2h[r] = bias2; c2l[r] = 0.f; }
    const long f2 = (long)nt * (H_DIM / 32);
    for (int kt = 0; kt < H_DIM / 32; ++kt) {
      v16bf a  = load_afrag(s1, H_DIM, kt, l15, khalf);
      v16bf bh = load_bfrag(w2h, f2 + kt, lane);
      v16bf bl = load_bfrag(w2l, f2 + kt, lane);
      if (kt + 1 < H_DIM / 32) {
        __builtin_prefetch((const char*)w2h + (f2 + kt + 1) * 512 + lane * 32, 0, 1);
        __builtin_prefetch((const char*)w2l + (f2 + kt + 1) * 512 + lane * 32, 0, 1);
      }
      c2h = wmma_bf16(a, bh, c2h);
      c2l = wmma_bf16(a, bl, c2l);
    }
    v8f c2 = c2h + c2l;
    // LIF 2; spike count stays in registers (readout is linear in s2)
#pragma unroll
    for (int r = 0; r < 8; ++r) {
      float v = BETA * v2[r] + c2[r];
      float s = (v > THR) ? 1.0f : 0.0f;
      v2[r] = v - s * THR;
      spkcnt[r] += s;
    }
    ++round;
    grid_barrier(bar, round * NBLK);            // all done reading this s1
  }

  // publish spike counts (integers <=128, exact in bf16)
#pragma unroll
  for (int r = 0; r < 8; ++r)
    acc2[(size_t)(r + khalf * 8) * H_DIM + nt * 16 + l15] =
        __float2bfloat16(spkcnt[r]);
  ++round;
  grid_barrier(bar, round * NBLK);

  // ---- layer 3 (once): out = acc2 @ W3^T + T*b3 -> waves 0..63 ------------
  if (nt < OUT_DIM / 16) {
    v8f c3h, c3l;
#pragma unroll
    for (int r = 0; r < 8; ++r) {
      c3h[r] = (float)T_STEPS * b3[nt * 16 + l15];
      c3l[r] = 0.f;
    }
    const long f3 = (long)nt * (H_DIM / 32);
    for (int kt = 0; kt < H_DIM / 32; ++kt) {
      v16bf a  = load_afrag(acc2, H_DIM, kt, l15, khalf);
      v16bf bh = load_bfrag(w3h, f3 + kt, lane);
      v16bf bl = load_bfrag(w3l, f3 + kt, lane);
      c3h = wmma_bf16(a, bh, c3h);
      c3l = wmma_bf16(a, bl, c3l);
    }
    v8f c3 = c3h + c3l;
#pragma unroll
    for (int r = 0; r < 8; ++r)
      out[(size_t)(r + khalf * 8) * OUT_DIM + nt * 16 + l15] = c3[r];
  }
}

// ---------------- host launch ----------------------------------------------

extern "C" void kernel_launch(void* const* d_in, const int* in_sizes, int n_in,
                              void* d_out, int out_size, void* d_ws,
                              size_t ws_size, hipStream_t stream) {
  const float* x  = (const float*)d_in[0];   // [16,128,1024]
  const float* w1 = (const float*)d_in[1];   // [4096,1024]
  const float* b1 = (const float*)d_in[2];   // [4096]
  const float* w2 = (const float*)d_in[3];   // [4096,4096]
  const float* b2 = (const float*)d_in[4];   // [4096]
  const float* w3 = (const float*)d_in[5];   // [1024,4096]
  const float* b3 = (const float*)d_in[6];   // [1024]
  float* out = (float*)d_out;                // [16,1024]

  char* ws = (char*)d_ws;
  unsigned* bar = (unsigned*)ws;             // 256B reserved
  const size_t XB = (size_t)T_STEPS * BATCH * IN_DIM;
  const size_t W1 = (size_t)H_DIM * IN_DIM;
  const size_t W2 = (size_t)H_DIM * H_DIM;
  const size_t W3 = (size_t)OUT_DIM * H_DIM;
  __hip_bfloat16* xh  = (__hip_bfloat16*)(ws + 256);
  __hip_bfloat16* xl  = xh + XB;
  __hip_bfloat16* w1h = xl + XB;
  __hip_bfloat16* w1l = w1h + W1;
  __hip_bfloat16* w2h = w1l + W1;
  __hip_bfloat16* w2l = w2h + W2;
  __hip_bfloat16* w3h = w2l + W2;
  __hip_bfloat16* w3l = w3h + W3;
  __hip_bfloat16* s1  = w3l + W3;
  __hip_bfloat16* ac2 = s1 + (size_t)BATCH * H_DIM;

  init_bar<<<1, 1, 0, stream>>>(bar);
  pack_x<<<(unsigned)((XB + 255) / 256), 256, 0, stream>>>(x, xh, xl);
  pack_w<<<(unsigned)((W1 / 2 + 255) / 256), 256, 0, stream>>>(w1, w1h, w1l, H_DIM, IN_DIM);
  pack_w<<<(unsigned)((W2 / 2 + 255) / 256), 256, 0, stream>>>(w2, w2h, w2l, H_DIM, H_DIM);
  pack_w<<<(unsigned)((W3 / 2 + 255) / 256), 256, 0, stream>>>(w3, w3h, w3l, OUT_DIM, H_DIM);
  snn_persistent<<<NBLK, TPB, 0, stream>>>(xh, xl, w1h, w1l, w2h, w2l, w3h, w3l,
                                           b1, b2, b3, s1, ac2, bar, out);
}
